// CopyNetDecoder_44813688767116
// MI455X (gfx1250) — matrix operand order repaired and checked by reading, not compiled, not run on hardware
//
#include <hip/hip_runtime.h>
#include <hip/hip_bf16.h>

// ---------------- problem constants ----------------
#define Bz   64
#define Sz   128
#define Hz   512
#define Ez   256
#define Vz   32000
#define Tz   32
#define EXT  (Vz + Sz)        // 32128
#define CAT  (Vz + EXT)       // 64128
#define XDIM (2 * Hz + Ez)    // 1280
#define G3H  (3 * Hz)         // 1536

// ---------------- WMMA fragment types ----------------
typedef __bf16 v16bf __attribute__((ext_vector_type(16)));
typedef __bf16 v8bf  __attribute__((ext_vector_type(8)));
typedef float  v8f   __attribute__((ext_vector_type(8)));
typedef unsigned int v4u __attribute__((ext_vector_type(4)));
typedef int v4i __attribute__((ext_vector_type(4)));
typedef int v8i __attribute__((ext_vector_type(8)));

union ABFrag { v16bf v; v8bf h[2]; };

__device__ __forceinline__ unsigned short f2bf(float f) {
    union { float f; unsigned u; } v; v.f = f;
    unsigned r = v.u + 0x7FFFu + ((v.u >> 16) & 1u);   // round-to-nearest-even
    return (unsigned short)(r >> 16);
}
__device__ __forceinline__ float sigm(float x) { return 1.f / (1.f + __expf(-x)); }

// ---- async global->LDS (ASYNCcnt) ----
__device__ __forceinline__ void stage_async8(void* dstLds, const void* srcGlob) {
    unsigned lds = (unsigned)(uintptr_t)dstLds;            // low 32 bits = LDS offset
    unsigned long long ga = (unsigned long long)(uintptr_t)srcGlob;
    asm volatile("global_load_async_to_lds_b64 %0, %1, off"
                 :: "v"(lds), "v"(ga) : "memory");
}
__device__ __forceinline__ void wait_async0() {
    asm volatile("s_wait_asynccnt 0x0" ::: "memory");
}

// ---- Tensor Data Mover: 2-D tile (bf16) global -> LDS, TENSORcnt-tracked ----
// D# per ISA 08_async_tensor §8.3/8.4: group0 = {count/type, lds_addr, global_addr},
// group1 = {data_size, tensor dims/strides, tile dims}; groups 2/3 zero (2-D tensor).
__device__ __forceinline__ void tdm_load_tile_bf16(
    unsigned ldsOff,            // LDS byte offset of tile destination
    const void* tileStart,      // global address of tile start
    int tensorRows,             // tensor_dim1 (rows)
    int tensorRowElems,         // tensor_dim0 and row stride, in 2B elements
    int tileRows,               // tile_dim1
    int tileElems)              // tile_dim0 (2B elements; *2 bytes must be %4==0)
{
    unsigned long long ga = (unsigned long long)(uintptr_t)tileStart;
    v4u g0;
    g0[0] = 1u;                                            // count=1, user descriptor
    g0[1] = ldsOff;                                        // lds_addr
    g0[2] = (unsigned)(ga & 0xFFFFFFFFu);                  // global_addr[31:0]
    g0[3] = (unsigned)((ga >> 32) & 0x01FFFFFFu) | (2u << 30);  // addr[56:32] | type=2
    v8i g1;
    g1[0] = (1 << 16);                                     // data_size=1 -> 2 bytes
    g1[1] = (tensorRowElems & 0xFFFF) << 16;               // tensor_dim0[15:0] @ bits 63:48
    g1[2] = ((tensorRowElems >> 16) & 0xFFFF)              // tensor_dim0[31:16]
          | ((tensorRows & 0xFFFF) << 16);                 // tensor_dim1[15:0]
    g1[3] = ((tensorRows >> 16) & 0xFFFF)                  // tensor_dim1[31:16]
          | ((tileElems & 0xFFFF) << 16);                  // tile_dim0
    g1[4] = (tileRows & 0xFFFF);                           // tile_dim1 (tile_dim2=0)
    g1[5] = tensorRowElems;                                // tensor_dim0_stride[31:0]
    g1[6] = 0;                                             // stride[47:32] | dim1_stride lo
    g1[7] = 0;
    v4i gz = {0, 0, 0, 0};
#if defined(__clang_major__) && (__clang_major__ >= 23)
    v8i gz8 = {0, 0, 0, 0, 0, 0, 0, 0};
    __builtin_amdgcn_tensor_load_to_lds(g0, g1, gz, gz, gz8, 0);
#else
    __builtin_amdgcn_tensor_load_to_lds(g0, g1, gz, gz, 0);
#endif
}

// =====================================================================
// WMMA GEMM:  C[64, N] = A[64, K](bf16) @ W[N, K](bf16)^T + bias
// Block = 8 waves covering a 64 x 64 panel at column blockIdx.x*64.
// wave: mtile = wave&3 (16-row slice), npair = wave>>2 (32-col slice);
// each wave keeps TWO 16x16 accumulators sharing one A fragment.
// A panel double-buffered in LDS, filled by the Tensor Data Mover
// (one tensor_load_to_lds per K-step, issued by wave 0 only).
// =====================================================================
__global__ void __launch_bounds__(256)
gemm64_wmma_bf16(const unsigned short* __restrict__ A,   // [64, K] bf16 bits
                 const unsigned short* __restrict__ W,   // [N, K] bf16 bits
                 const float* __restrict__ bias,         // [N]
                 float* __restrict__ C,                  // [64, ldc]
                 int N, int K, int ldc)
{
    __shared__ __align__(32) unsigned short As[2][64 * 32];

    const int tid   = threadIdx.x;
    const int wave  = tid >> 5;
    const int lane  = tid & 31;
    const int hf    = lane >> 4;          // wave half: 0 or 1
    const int r     = lane & 15;
    const int mtile = wave & 3;           // rows 16*mtile .. +15
    const int npair = wave >> 2;          // 0..1 -> 32-col slice
    const int n0    = blockIdx.x * 64 + npair * 32;
    const bool leader = (wave == 0);      // TDM issuing wave

    const unsigned short* wrow0 = W + (size_t)(n0 + r) * K;
    const unsigned short* wrow1 = W + (size_t)(n0 + 16 + r) * K;

    if (leader)                           // prologue: tile for kk = 0
        tdm_load_tile_bf16((unsigned)(uintptr_t)&As[0][0], A, 64, K, 64, 32);

    v8f acc0 = {}, acc1 = {};
    for (int kk = 0; kk < K; kk += 32) {
        const int cur = (kk >> 5) & 1;
        if (leader) __builtin_amdgcn_s_wait_tensorcnt(0);  // DMA landed in LDS
        __syncthreads();                                   // visible to all waves
        if (kk + 32 < K) {
            if (leader)
                tdm_load_tile_bf16((unsigned)(uintptr_t)&As[cur ^ 1][0],
                                   A + kk + 32, 64, K, 64, 32);
            __builtin_prefetch(wrow0 + kk + 32, 0, 1);     // global_prefetch_b8
        }

        // A fragment (16x32): lanes 0-15 K{0..15}, lanes 16-31 K{16..31}
        ABFrag a;
        const int m = mtile * 16 + r;
        a.h[0] = *(const v8bf*)&As[cur][m * 32 + 16 * hf];
        a.h[1] = *(const v8bf*)&As[cur][m * 32 + 16 * hf + 8];

        // B fragments (32x16): lane col = r, K = e + 16*hf -> contiguous 32B loads
        v16bf b0 = *(const v16bf*)&wrow0[kk + 16 * hf];
        v16bf b1 = *(const v16bf*)&wrow1[kk + 16 * hf];

        acc0 = __builtin_amdgcn_wmma_f32_16x16x32_bf16(
            false, a.v, false, b0, (short)0, acc0, false, false);
        acc1 = __builtin_amdgcn_wmma_f32_16x16x32_bf16(
            false, a.v, false, b1, (short)0, acc1, false, false);
    }

    const float b0s = bias ? bias[n0 + r] : 0.f;
    const float b1s = bias ? bias[n0 + 16 + r] : 0.f;
#pragma unroll
    for (int e = 0; e < 8; ++e) {
        const int m = mtile * 16 + 8 * hf + e;           // C: VGPR e -> M = e + 8*half
        C[(size_t)m * ldc + n0 + r]      = acc0[e] + b0s;
        C[(size_t)m * ldc + n0 + 16 + r] = acc1[e] + b1s;
    }
}

// =====================================================================
// elementwise / reduction helper kernels
// =====================================================================
__global__ void cvt_f32_bf16(const float* __restrict__ src,
                             unsigned short* __restrict__ dst, int n) {
    int i = blockIdx.x * blockDim.x + threadIdx.x;
    if (i < n) dst[i] = f2bf(src[i]);
}

// scores[b,s] = dot(enc[b,s,:], th[b,:]) — one wave per (b,s); th row (shared
// by all 8 waves of the block) staged into LDS with async-to-LDS loads.
__global__ void __launch_bounds__(256)
bsh_bh_bs(const float* __restrict__ enc, const float* __restrict__ th,
          float* __restrict__ out) {
    __shared__ __align__(16) float tsh[Hz];
    int wave = threadIdx.x >> 5, lane = threadIdx.x & 31;
    int g = blockIdx.x * 8 + wave;          // 8 | 128 -> whole block shares b
    int b = g >> 7, s = g & (Sz - 1);
    {
        int i = threadIdx.x * 2;            // 256 threads x 8B = 2KB row
        stage_async8(&tsh[i], th + (size_t)b * Hz + i);
    }
    wait_async0();
    __syncthreads();
    const float* e = enc + ((size_t)b * Sz + s) * Hz;
    float acc = 0.f;
    for (int k = lane; k < Hz; k += 32) acc += e[k] * tsh[k];
    for (int off = 16; off; off >>= 1) acc += __shfl_xor(acc, off, 32);
    if (lane == 0) out[g] = acc;
}

__global__ void __launch_bounds__(128)
softmax_s(const float* __restrict__ sc, float* __restrict__ aw) {
    int b = blockIdx.x, tid = threadIdx.x;
    __shared__ float red[128];
    float v = sc[b * Sz + tid];
    red[tid] = v; __syncthreads();
    for (int o = 64; o; o >>= 1) { if (tid < o) red[tid] = fmaxf(red[tid], red[tid + o]); __syncthreads(); }
    float mx = red[0]; __syncthreads();
    float ev = __expf(v - mx);
    red[tid] = ev; __syncthreads();
    for (int o = 64; o; o >>= 1) { if (tid < o) red[tid] += red[tid + o]; __syncthreads(); }
    aw[b * Sz + tid] = ev / red[0];
}

__global__ void __launch_bounds__(256)
ctx_kernel(const float* __restrict__ aw, const float* __restrict__ enc,
           float* __restrict__ ctx) {
    int b = blockIdx.x;
    __shared__ float a[Sz];
    if (threadIdx.x < Sz) a[threadIdx.x] = aw[b * Sz + threadIdx.x];
    __syncthreads();
    for (int h = threadIdx.x; h < Hz; h += blockDim.x) {
        float acc = 0.f;
        const float* e = enc + (size_t)b * Sz * Hz + h;
        for (int s = 0; s < Sz; ++s) acc += a[s] * e[s * Hz];
        ctx[b * Hz + h] = acc;
    }
}

__global__ void __launch_bounds__(256)
build_x(const float* __restrict__ ctx, const float* __restrict__ sel,
        const float* __restrict__ emb_w, const int* __restrict__ idx,
        unsigned short* __restrict__ xbf) {
    int b = blockIdx.x;
    int id = idx[b];
    int idc = (id > Vz) ? 3 : id;
    if (idc >= Vz) idc = Vz - 1;           // JAX OOB-gather clamp
    for (int j = threadIdx.x; j < XDIM; j += blockDim.x) {
        float v = (j < Hz) ? ctx[b * Hz + j]
                : (j < 2 * Hz) ? sel[b * Hz + (j - Hz)]
                : emb_w[(size_t)idc * Ez + (j - 2 * Hz)];
        xbf[b * XDIM + j] = f2bf(v);
    }
}

__global__ void __launch_bounds__(256)
gru_combine(const float* __restrict__ gi, const float* __restrict__ gh,
            const float* __restrict__ hid, float* __restrict__ hn,
            unsigned short* __restrict__ hnbf) {
    int i = blockIdx.x * blockDim.x + threadIdx.x;
    if (i >= Bz * Hz) return;
    int b = i >> 9, h = i & (Hz - 1);
    const float* gib = gi + (size_t)b * G3H;
    const float* ghb = gh + (size_t)b * G3H;
    float r = sigm(gib[h] + ghb[h]);
    float z = sigm(gib[Hz + h] + ghb[Hz + h]);
    float n = tanhf(gib[2 * Hz + h] + r * ghb[2 * Hz + h]);
    float o = (1.f - z) * n + z * hid[i];
    hn[i] = o; hnbf[i] = f2bf(o);
}

__global__ void copy_fill(const unsigned char* __restrict__ missing,
                          float* __restrict__ logits) {
    long i = (long)blockIdx.x * blockDim.x + threadIdx.x;
    if (i >= (long)Bz * EXT) return;
    int b = (int)(i / EXT), j = (int)(i % EXT);
    logits[(size_t)b * CAT + Vz + j] = missing[i] ? -1e6f : 0.f;
}

__global__ void copy_scatter(const int* __restrict__ inputs,
                             const float* __restrict__ css,
                             float* __restrict__ logits) {
    int i = blockIdx.x * blockDim.x + threadIdx.x;
    if (i >= Bz * Sz) return;
    int b = i >> 7;
    atomicAdd(&logits[(size_t)b * CAT + Vz + inputs[i]], css[i]);
}

__global__ void fixup_logits(float* __restrict__ logits) {
    int b = threadIdx.x;
    if (b < Bz) {
        logits[(size_t)b * CAT] = -1e6f;        // gen[:,0]
        logits[(size_t)b * CAT + Vz] = -1e6f;   // copy ext position 0
    }
}

__global__ void __launch_bounds__(256)
softmax_cat(const float* __restrict__ logits, float* __restrict__ rowmax,
            float* __restrict__ rowsum) {
    int b = blockIdx.x, tid = threadIdx.x;
    __shared__ float red[256];
    const float* L = logits + (size_t)b * CAT;
    float mx = -INFINITY;
    for (int j = tid; j < CAT; j += 256) mx = fmaxf(mx, L[j]);
    red[tid] = mx; __syncthreads();
    for (int o = 128; o; o >>= 1) { if (tid < o) red[tid] = fmaxf(red[tid], red[tid + o]); __syncthreads(); }
    mx = red[0]; __syncthreads();
    float sm = 0.f;
    for (int j = tid; j < CAT; j += 256) sm += __expf(L[j] - mx);
    red[tid] = sm; __syncthreads();
    for (int o = 128; o; o >>= 1) { if (tid < o) red[tid] += red[tid + o]; __syncthreads(); }
    if (tid == 0) { rowmax[b] = mx; rowsum[b] = red[0]; }
}

__global__ void __launch_bounds__(256)
finalize_step(const float* __restrict__ logits, const float* __restrict__ rowmax,
              const float* __restrict__ rowsum, float* __restrict__ dec_out,
              float* __restrict__ samp_out, int* __restrict__ idx, int t) {
    int b = blockIdx.x, tid = threadIdx.x;
    __shared__ float bred[256];
    __shared__ int   bidx[256];
    const float* L = logits + (size_t)b * CAT;
    float mx = rowmax[b], inv = 1.f / rowsum[b];
    float best = -INFINITY; int bj = 0;
    float* orow = dec_out + ((size_t)b * Tz + t) * EXT;
    for (int j = tid; j < EXT; j += 256) {
        float p = (j < Vz)
            ? (__expf(L[j] - mx) + __expf(L[Vz + j] - mx)) * inv
            : __expf(L[Vz + j] - mx) * inv;
        float lp = __logf(p + 1e-10f);
        orow[j] = lp;
        if (lp > best) { best = lp; bj = j; }
    }
    bred[tid] = best; bidx[tid] = bj; __syncthreads();
    for (int o = 128; o; o >>= 1) {
        if (tid < o) {
            if (bred[tid + o] > bred[tid] ||
                (bred[tid + o] == bred[tid] && bidx[tid + o] < bidx[tid])) {
                bred[tid] = bred[tid + o]; bidx[tid] = bidx[tid + o];
            }
        }
        __syncthreads();
    }
    if (tid == 0) { idx[b] = bidx[0]; samp_out[(size_t)b * Tz + t] = (float)bidx[0]; }
}

__global__ void __launch_bounds__(256)
sel_update(const int* __restrict__ inputs, const float* __restrict__ css,
           const float* __restrict__ enc, const int* __restrict__ idx,
           float* __restrict__ sel) {
    int b = blockIdx.x, tid = threadIdx.x;
    __shared__ float ss[Sz];
    __shared__ float red[256];
    int ni = idx[b];
    float loc = 0.f;
    for (int s = tid; s < Sz; s += 256) {
        float v = (inputs[b * Sz + s] == ni) ? css[b * Sz + s] : 0.f;
        ss[s] = v; loc += fabsf(v);
    }
    red[tid] = loc; __syncthreads();
    for (int o = 128; o; o >>= 1) { if (tid < o) red[tid] += red[tid + o]; __syncthreads(); }
    float norm = fmaxf(red[0], 1e-12f);
    __syncthreads();
    for (int h = tid; h < Hz; h += 256) {
        float acc = 0.f;
        const float* e = enc + (size_t)b * Sz * Hz + h;
        for (int s = 0; s < Sz; ++s) acc += ss[s] * e[s * Hz];
        sel[b * Hz + h] = acc / norm;
    }
}

__global__ void init_state(float* hid, unsigned short* hidbf, float* sel, int* idx) {
    int i = blockIdx.x * blockDim.x + threadIdx.x;
    if (i < Bz * Hz) { hid[i] = 0.f; hidbf[i] = 0; sel[i] = 0.f; }
    if (i < Bz) idx[i] = 1;
}

__global__ void init_out(float* dec_out, float* samp) {
    long i = (long)blockIdx.x * blockDim.x + threadIdx.x;
    if (i >= (long)Bz * EXT) return;
    int b = (int)(i / EXT), j = (int)(i % EXT);
    dec_out[(size_t)b * Tz * EXT + j] = (j == 1) ? 1.f : 0.f;
    if (j == 0) samp[(size_t)b * Tz] = 1.f;
}

__global__ void missing_init(unsigned char* m) {
    long i = (long)blockIdx.x * blockDim.x + threadIdx.x;
    if (i < (long)Bz * EXT) m[i] = 1;
}
__global__ void missing_mark(const int* __restrict__ inputs, unsigned char* m) {
    int i = blockIdx.x * blockDim.x + threadIdx.x;
    if (i >= Bz * Sz) return;
    int b = i >> 7;
    m[(size_t)b * EXT + inputs[i]] = 0;
}
__global__ void missing_fix(unsigned char* m) {
    int b = threadIdx.x;
    if (b < Bz) m[(size_t)b * EXT] = 1;
}

// =====================================================================
// host orchestration
// =====================================================================
static inline char* wsalloc(char*& p, size_t bytes) {
    char* r = p;
    p += (bytes + 255) & ~(size_t)255;
    return r;
}

extern "C" void kernel_launch(void* const* d_in, const int* in_sizes, int n_in,
                              void* d_out, int out_size, void* d_ws, size_t ws_size,
                              hipStream_t stream) {
    const float* enc     = (const float*)d_in[0];   // [B,S,H]
    const int*   inputs  = (const int*)d_in[1];     // [B,S]
    const float* emb_w   = (const float*)d_in[3];   // [V,E]
    const float* attn_w  = (const float*)d_in[4];
    const float* attn_b  = (const float*)d_in[5];
    const float* copy_w  = (const float*)d_in[6];
    const float* copy_b  = (const float*)d_in[7];
    const float* gru_wih = (const float*)d_in[8];
    const float* gru_whh = (const float*)d_in[9];
    const float* gru_bih = (const float*)d_in[10];
    const float* gru_bhh = (const float*)d_in[11];
    const float* out_w   = (const float*)d_in[12];
    const float* out_b   = (const float*)d_in[13];

    float* dec_out = (float*)d_out;                       // [B,T,EXT]
    float* samp    = dec_out + (size_t)Bz * Tz * EXT;     // [B,T]

    char* p = (char*)d_ws;
    unsigned short* ow_bf  = (unsigned short*)wsalloc(p, sizeof(short) * (size_t)Vz * Hz);
    unsigned short* wih_bf = (unsigned short*)wsalloc(p, sizeof(short) * (size_t)G3H * XDIM);
    unsigned short* whh_bf = (unsigned short*)wsalloc(p, sizeof(short) * (size_t)G3H * Hz);
    unsigned short* aw_bf  = (unsigned short*)wsalloc(p, sizeof(short) * (size_t)Hz * Hz);
    unsigned short* cw_bf  = (unsigned short*)wsalloc(p, sizeof(short) * (size_t)Hz * Hz);
    float* logits  = (float*)wsalloc(p, sizeof(float) * (size_t)Bz * CAT);
    unsigned char* missing = (unsigned char*)wsalloc(p, (size_t)Bz * EXT);
    float* hA      = (float*)wsalloc(p, sizeof(float) * Bz * Hz);
    float* hB      = (float*)wsalloc(p, sizeof(float) * Bz * Hz);
    unsigned short* hAbf = (unsigned short*)wsalloc(p, sizeof(short) * Bz * Hz);
    unsigned short* hBbf = (unsigned short*)wsalloc(p, sizeof(short) * Bz * Hz);
    float* sel     = (float*)wsalloc(p, sizeof(float) * Bz * Hz);
    float* th      = (float*)wsalloc(p, sizeof(float) * Bz * Hz);
    float* th2     = (float*)wsalloc(p, sizeof(float) * Bz * Hz);
    float* ctx     = (float*)wsalloc(p, sizeof(float) * Bz * Hz);
    unsigned short* xbf = (unsigned short*)wsalloc(p, sizeof(short) * Bz * XDIM);
    float* gi      = (float*)wsalloc(p, sizeof(float) * Bz * G3H);
    float* gh      = (float*)wsalloc(p, sizeof(float) * Bz * G3H);
    float* scores  = (float*)wsalloc(p, sizeof(float) * Bz * Sz);
    float* awgt    = (float*)wsalloc(p, sizeof(float) * Bz * Sz);
    float* css     = (float*)wsalloc(p, sizeof(float) * Bz * Sz);
    float* rowmax  = (float*)wsalloc(p, sizeof(float) * Bz);
    float* rowsum  = (float*)wsalloc(p, sizeof(float) * Bz);
    int*   idx     = (int*)wsalloc(p, sizeof(int) * Bz);

    auto cdiv = [](long a, long b) { return (int)((a + b - 1) / b); };

    // weights -> bf16 (once per call)
    cvt_f32_bf16<<<cdiv((long)Vz * Hz, 256), 256, 0, stream>>>(out_w, ow_bf, Vz * Hz);
    cvt_f32_bf16<<<cdiv((long)G3H * XDIM, 256), 256, 0, stream>>>(gru_wih, wih_bf, G3H * XDIM);
    cvt_f32_bf16<<<cdiv((long)G3H * Hz, 256), 256, 0, stream>>>(gru_whh, whh_bf, G3H * Hz);
    cvt_f32_bf16<<<cdiv((long)Hz * Hz, 256), 256, 0, stream>>>(attn_w, aw_bf, Hz * Hz);
    cvt_f32_bf16<<<cdiv((long)Hz * Hz, 256), 256, 0, stream>>>(copy_w, cw_bf, Hz * Hz);

    // token-presence mask + state init + t=0 outputs
    missing_init<<<cdiv((long)Bz * EXT, 256), 256, 0, stream>>>(missing);
    missing_mark<<<cdiv(Bz * Sz, 256), 256, 0, stream>>>(inputs, missing);
    missing_fix<<<1, 64, 0, stream>>>(missing);
    init_state<<<cdiv(Bz * Hz, 256), 256, 0, stream>>>(hA, hAbf, sel, idx);
    init_out<<<cdiv((long)Bz * EXT, 256), 256, 0, stream>>>(dec_out, samp);

    const float* hid = hA;       unsigned short* hidbf = hAbf;
    float* hnew = hB;            unsigned short* hnewbf = hBbf;

    for (int t = 1; t < Tz; ++t) {
        // th = hid @ attn_w^T + attn_b
        gemm64_wmma_bf16<<<Hz / 64, 256, 0, stream>>>(hidbf, aw_bf, attn_b, th, Hz, Hz, Hz);
        // attention scores -> softmax -> context
        bsh_bh_bs<<<Bz * Sz / 8, 256, 0, stream>>>(enc, th, scores);
        softmax_s<<<Bz, 128, 0, stream>>>(scores, awgt);
        ctx_kernel<<<Bz, 256, 0, stream>>>(awgt, enc, ctx);
        // rnn input = [context, sel, emb]
        build_x<<<Bz, 256, 0, stream>>>(ctx, sel, emb_w, idx, xbf);
        // GRU gates
        gemm64_wmma_bf16<<<G3H / 64, 256, 0, stream>>>(xbf, wih_bf, gru_bih, gi, G3H, XDIM, G3H);
        gemm64_wmma_bf16<<<G3H / 64, 256, 0, stream>>>(hidbf, whh_bf, gru_bhh, gh, G3H, Hz, G3H);
        gru_combine<<<cdiv(Bz * Hz, 256), 256, 0, stream>>>(gi, gh, hid, hnew, hnewbf);
        // copy scores
        gemm64_wmma_bf16<<<Hz / 64, 256, 0, stream>>>(hnewbf, cw_bf, copy_b, th2, Hz, Hz, Hz);
        bsh_bh_bs<<<Bz * Sz / 8, 256, 0, stream>>>(enc, th2, css);
        // generation logits (big GEMM, L2-resident bf16 weights)
        gemm64_wmma_bf16<<<Vz / 64, 256, 0, stream>>>(hnewbf, ow_bf, out_b, logits, Vz, Hz, CAT);
        // extended-vocab copy region
        copy_fill<<<cdiv((long)Bz * EXT, 256), 256, 0, stream>>>(missing, logits);
        copy_scatter<<<cdiv(Bz * Sz, 256), 256, 0, stream>>>(inputs, css, logits);
        fixup_logits<<<1, 64, 0, stream>>>(logits);
        // softmax over [V + EXT], fold copy-of-vocab, logp, argmax
        softmax_cat<<<Bz, 256, 0, stream>>>(logits, rowmax, rowsum);
        finalize_step<<<Bz, 256, 0, stream>>>(logits, rowmax, rowsum, dec_out, samp, idx, t);
        // selective read
        sel_update<<<Bz, 256, 0, stream>>>(inputs, css, enc, idx, sel);

        // ping-pong hidden state
        const float* tmpc = hid; hid = hnew; hnew = (float*)tmpc;
        unsigned short* tmpb = hidbf; hidbf = hnewbf; hnewbf = tmpb;
    }
    (void)in_sizes; (void)n_in; (void)out_size; (void)ws_size;
}
